// ContextEncoder_37950331027524
// MI455X (gfx1250) — matrix operand order repaired and checked by reading, not compiled, not run on hardware
//
#include <hip/hip_runtime.h>
#include <math.h>
#include <stdint.h>

typedef __attribute__((ext_vector_type(16))) _Float16 v16h;
typedef __attribute__((ext_vector_type(8)))  float    v8f;

static constexpr int B = 256;
static constexpr int S = 128;
static constexpr int D = 512;
static constexpr int H = 512;

#define EPSF 1e-15f
#define MAXT (1.0f - 1e-5f)

// ---------------------------------------------------------------- reductions
__device__ __forceinline__ float block_sum(float v) {
  __shared__ float sred[256];
  const int t = threadIdx.x;
  sred[t] = v;
  __syncthreads();
#pragma unroll
  for (int o = 128; o > 0; o >>= 1) {
    if (t < o) sred[t] += sred[t + o];
    __syncthreads();
  }
  float r = sred[0];
  __syncthreads();
  return r;
}

__device__ __forceinline__ float clipf(float x, float lo, float hi) {
  return fminf(fmaxf(x, lo), hi);
}

// ---------------------------------------------------------------- setup kernels
__global__ __launch_bounds__(256) void gather_embed(
    const int* __restrict__ ctx, const float* __restrict__ lut,
    _Float16* __restrict__ xh16, float* __restrict__ xnorm) {
  const int row = blockIdx.x;            // b*S + s
  const int idx = ctx[row];
  const float* src = lut + (long)idx * D;
  float p = 0.f;
  for (int i = threadIdx.x; i < D; i += 256) {
    float v = src[i];
    xh16[(long)row * D + i] = (_Float16)v;
    p += v * v;
  }
  float tot = block_sum(p);
  if (threadIdx.x == 0) xnorm[row] = sqrtf(tot + 1e-30f);
}

__global__ void f32_to_f16(const float* __restrict__ src,
                           _Float16* __restrict__ dst, int n) {
  int i = blockIdx.x * blockDim.x + threadIdx.x;
  if (i < n) dst[i] = (_Float16)src[i];
}

__global__ void init_state(float* h, _Float16* h16, float* hn) {
  int i = blockIdx.x * blockDim.x + threadIdx.x;
  if (i < B * H) { h[i] = 0.f; h16[i] = (_Float16)0.f; }
  if (i < B) hn[i] = sqrtf(1e-30f);
}

// ---------------------------------------------------------------- WMMA GEMM
// C[M,N] = A[M,K] @ W[N,K]^T ; A,W f16 row-major, C f32 row-major.
// One 16x16 C tile per wave32; 4 waves/block stacked along M (block tile 64x16).
//  * B slab (16 x K, shared by the 4 waves) async-staged to LDS once.
//  * A tiles (16 x 32 per wave) double-buffered through wave-private LDS with
//    GLOBAL_LOAD_ASYNC_TO_LDS_B128; s_wait_asynccnt 0x2 = "current buffer done,
//    next buffer still in flight" (ASYNCcnt completes in order) -> the global
//    A stream is fully pipelined behind the WMMA chain, no loadcnt-0 stalls.
__global__ __launch_bounds__(128) void wmma_gemm(
    const _Float16* __restrict__ A, int lda,
    const _Float16* __restrict__ W, int ldw,
    float* __restrict__ C, int ldc, int K) {
  __shared__ _Float16 Bs[16][520];        // 512 + 8 pad: conflict-free rows
  __shared__ _Float16 As[4][2][16][40];   // [wave][buf][row][32 + 8 pad]
  const int tid  = threadIdx.x;
  const int lane = tid & 31;
  const int wave = tid >> 5;
  const int tn   = blockIdx.x * 16;
  const int tm   = (blockIdx.y * 4 + wave) * 16;
  const int half = lane >> 4;       // 0 | 1
  const int lrow = lane & 15;

  // ---- async-stage B slab: rows tn..tn+15, K halfs each, 16B per transfer
  const int cpr   = K >> 3;         // 16B chunks per row (K=512 -> 64)
  const int total = 16 * cpr;       // 1024 chunks
  for (int c = tid; c < total; c += 128) {
    const int n  = c / cpr;
    const int kc = (c % cpr) << 3;
    unsigned long long gaddr =
        (unsigned long long)(uintptr_t)(W + (size_t)(tn + n) * ldw + kc);
    unsigned laddr = (unsigned)(uintptr_t)&Bs[n][kc];  // low 32b = LDS offset
    asm volatile("global_load_async_to_lds_b128 %0, %1, off"
                 :: "v"(laddr), "v"(gaddr) : "memory");
  }

  // ---- wave-private A-tile stage: 16 rows x 32 halfs = 64 x 16B chunks,
  //      each lane issues 2 async transfers.
  auto stage_a = [&](int k0, int buf) {
#pragma unroll
    for (int i = 0; i < 2; ++i) {
      const int c    = lane + i * 32;   // 0..63
      const int n    = c >> 2;          // row in tile
      const int koff = (c & 3) << 3;    // 0,8,16,24 halfs
      unsigned long long gaddr = (unsigned long long)(uintptr_t)(
          A + (size_t)(tm + n) * lda + (k0 + koff));
      unsigned laddr = (unsigned)(uintptr_t)&As[wave][buf][n][koff];
      asm volatile("global_load_async_to_lds_b128 %0, %1, off"
                   :: "v"(laddr), "v"(gaddr) : "memory");
    }
  };

  stage_a(0, 0);
  asm volatile("s_wait_asynccnt 0x0" ::: "memory");  // drain B slab + A[0]
  __syncthreads();                                   // B visible to all waves

  v8f acc = {};
  for (int k0 = 0; k0 < K; k0 += 32) {
    const int buf = (k0 >> 5) & 1;
    if (k0 + 32 < K) {
      stage_a(k0 + 32, buf ^ 1);                     // 2 newest in flight
      asm volatile("s_wait_asynccnt 0x2" ::: "memory");
    } else {
      asm volatile("s_wait_asynccnt 0x0" ::: "memory");
    }
    // A fragment (16x32 ISA layout) from wave-private LDS buffer
    v16h af = {};
#pragma unroll
    for (int v = 0; v < 8; ++v) {
      int ka = ((v < 4) ? 2 * v : 2 * v + 8) + half * 8;
      af[2 * v]     = As[wave][buf][lrow][ka];
      af[2 * v + 1] = As[wave][buf][lrow][ka + 1];
    }
    // B fragment (32x16): n = lane&15, same K order as A (K=8h+s, s<8; 16+8h+s-8, s>=8)
    v16h bf = {};
#pragma unroll
    for (int v = 0; v < 8; ++v) {
      int kb = ((v < 4) ? 2 * v : 2 * v + 8) + half * 8 + k0;
      bf[2 * v]     = Bs[lrow][kb];
      bf[2 * v + 1] = Bs[lrow][kb + 1];
    }
    acc = __builtin_amdgcn_wmma_f32_16x16x32_f16(
        /*neg_a=*/false, af, /*neg_b=*/false, bf,
        /*c_mod=*/(short)0, acc, /*reuse_a=*/false, /*reuse_b=*/false);
  }
  // C/D layout: VGPR r -> M = r + 8*half, N = lane&15
  const int n  = tn + lrow;
  const int mb = tm + half * 8;
#pragma unroll
  for (int r = 0; r < 8; ++r)
    C[(long)(mb + r) * ldc + n] = acc[r];
}

// ---------------------------------------------------------------- cell math
__device__ __forceinline__ void gate_pair(
    float m0, float m1, float srcn,   // raw W@h elems + ||h||
    float u0, float u1, float xn,     // raw U@x elems + ||x||
    float bb0, float bb1,             // bias elems
    float& g0, float& g1) {
  float mx2 = block_sum(m0 * m0 + m1 * m1);
  float mxn = fmaxf(sqrtf(mx2 + 1e-30f), EPSF);
  float nc  = clipf(srcn, EPSF, MAXT);
  float sW  = tanhf(mxn / nc * atanhf(nc)) / mxn;
  float w0 = sW * m0, w1 = sW * m1;
  float ux2 = block_sum(u0 * u0 + u1 * u1);
  float uxn = fmaxf(sqrtf(ux2 + 1e-30f), EPSF);
  float xc  = clipf(xn, EPSF, MAXT);
  float sU  = tanhf(uxn / xc * atanhf(xc)) / uxn;
  float y0 = sU * u0, y1 = sU * u1;
  float x2 = block_sum(w0 * w0 + w1 * w1);
  float y2 = block_sum(y0 * y0 + y1 * y1);
  float xy = block_sum(w0 * y0 + w1 * y1);
  float den = fmaxf(1.f + 2.f * xy + x2 * y2, EPSF);
  float cA = (1.f + 2.f * xy + y2), cB = (1.f - x2);
  float a0 = (cA * w0 + cB * y0) / den;
  float a1 = (cA * w1 + cB * y1) / den;
  float ax2 = block_sum(a0 * a0 + a1 * a1);
  float b2  = block_sum(bb0 * bb0 + bb1 * bb1);
  float ab  = block_sum(a0 * bb0 + a1 * bb1);
  float den2 = fmaxf(1.f + 2.f * ab + ax2 * b2, EPSF);
  float dA = (1.f + 2.f * ab + b2), dB = (1.f - ax2);
  float c0 = (dA * a0 + dB * bb0) / den2;
  float c1 = (dA * a1 + dB * bb1) / den2;
  float cn2 = block_sum(c0 * c0 + c1 * c1);
  float cn  = fmaxf(sqrtf(cn2 + 1e-30f), EPSF);
  float sc  = atanhf(clipf(cn, -MAXT, MAXT)) / cn;
  g0 = 1.f / (1.f + expf(-sc * c0));
  g1 = 1.f / (1.f + expf(-sc * c1));
}

// Stage A: z, r gates, rh = mobius_pointwise_mul(r, h); emits rh (f16) for GEMM2.
__global__ __launch_bounds__(256) void cell_gates(
    const float* __restrict__ hbuf, const float* __restrict__ hnorm,
    const float* __restrict__ wh_raw,           // [B,3H] (r@0, z@2H valid)
    const float* __restrict__ ux_all,           // [B*S,3H]
    const float* __restrict__ xnorm, const float* __restrict__ bias,
    int s,
    float* __restrict__ zbuf, _Float16* __restrict__ rh16,
    float* __restrict__ rhnorm) {
  const int b = blockIdx.x, t = threadIdx.x;
  const int i0 = t, i1 = t + 256;
  const float* hrow = hbuf + (long)b * H;
  const float h0 = hrow[i0], h1 = hrow[i1];
  const float hn = hnorm[b];
  const long  xr = (long)b * S + s;
  const float xn = xnorm[xr];
  const float* ux = ux_all + xr * (3 * H);
  const float* wh = wh_raw + (long)b * (3 * H);

  float z0, z1, r0, r1;
  gate_pair(wh[2 * H + i0], wh[2 * H + i1], hn,
            ux[2 * H + i0], ux[2 * H + i1], xn,
            bias[2 * H + i0], bias[2 * H + i1], z0, z1);     // z gate
  gate_pair(wh[i0], wh[i1], hn, ux[i0], ux[i1], xn,
            bias[i0], bias[i1], r0, r1);                     // r gate

  // rh = expmap0(r * logmap0(h))
  float hnm = fmaxf(hn, EPSF);
  float la  = atanhf(clipf(hnm, -MAXT, MAXT)) / hnm;
  float v0 = r0 * la * h0, v1 = r1 * la * h1;
  float nv2 = block_sum(v0 * v0 + v1 * v1);
  float nv  = fmaxf(sqrtf(nv2 + 1e-30f), EPSF);
  float se  = tanhf(nv) / nv;
  float rh0 = se * v0, rh1 = se * v1;

  zbuf[(long)b * H + i0] = z0; zbuf[(long)b * H + i1] = z1;
  rh16[(long)b * H + i0] = (_Float16)rh0;
  rh16[(long)b * H + i1] = (_Float16)rh1;
  float rn2 = block_sum(rh0 * rh0 + rh1 * rh1);
  if (t == 0) rhnorm[b] = sqrtf(rn2 + 1e-30f);
}

// Stage B: h_tilde, delta, h_new; writes state + output slice.
__global__ __launch_bounds__(256) void cell_update(
    float* hbuf, float* hnorm,                   // read then overwritten
    const float* __restrict__ wh_raw,            // hh gate at offset H (W_hhh@rh)
    const float* __restrict__ ux_all, const float* __restrict__ xnorm,
    const float* __restrict__ bias,
    const float* __restrict__ zbuf, const float* __restrict__ rhnorm,
    int s, _Float16* __restrict__ h16, float* __restrict__ out) {
  const int b = blockIdx.x, t = threadIdx.x;
  const int i0 = t, i1 = t + 256;
  const float h0 = hbuf[(long)b * H + i0], h1 = hbuf[(long)b * H + i1];
  const long  xr = (long)b * S + s;
  const float xn = xnorm[xr];
  const float* ux = ux_all + xr * (3 * H);
  const float* wh = wh_raw + (long)b * (3 * H);

  // h_tilde = one_rnn_transform(W_hhh, rh, W_ihh, x, b_h)
  float m0 = wh[H + i0], m1 = wh[H + i1];
  float mx2 = block_sum(m0 * m0 + m1 * m1);
  float mxn = fmaxf(sqrtf(mx2 + 1e-30f), EPSF);
  float rn  = clipf(rhnorm[b], EPSF, MAXT);
  float sW  = tanhf(mxn / rn * atanhf(rn)) / mxn;
  float w0 = sW * m0, w1 = sW * m1;

  float u0 = ux[H + i0], u1 = ux[H + i1];
  float ux2 = block_sum(u0 * u0 + u1 * u1);
  float uxn = fmaxf(sqrtf(ux2 + 1e-30f), EPSF);
  float xc  = clipf(xn, EPSF, MAXT);
  float sU  = tanhf(uxn / xc * atanhf(xc)) / uxn;
  float y0 = sU * u0, y1 = sU * u1;

  float x2 = block_sum(w0 * w0 + w1 * w1);
  float y2 = block_sum(y0 * y0 + y1 * y1);
  float xy = block_sum(w0 * y0 + w1 * y1);
  float den = fmaxf(1.f + 2.f * xy + x2 * y2, EPSF);
  float a0 = ((1.f + 2.f * xy + y2) * w0 + (1.f - x2) * y0) / den;
  float a1 = ((1.f + 2.f * xy + y2) * w1 + (1.f - x2) * y1) / den;

  float bb0 = bias[H + i0], bb1 = bias[H + i1];   // b_h
  float ax2 = block_sum(a0 * a0 + a1 * a1);
  float b2  = block_sum(bb0 * bb0 + bb1 * bb1);
  float ab  = block_sum(a0 * bb0 + a1 * bb1);
  float den2 = fmaxf(1.f + 2.f * ab + ax2 * b2, EPSF);
  float ht0 = ((1.f + 2.f * ab + b2) * a0 + (1.f - ax2) * bb0) / den2;
  float ht1 = ((1.f + 2.f * ab + b2) * a1 + (1.f - ax2) * bb1) / den2;

  // delta = mobius_add(-h, h_tilde)
  float nh0 = -h0, nh1 = -h1;
  float hx2 = block_sum(nh0 * nh0 + nh1 * nh1);
  float ty2 = block_sum(ht0 * ht0 + ht1 * ht1);
  float txy = block_sum(nh0 * ht0 + nh1 * ht1);
  float dden = fmaxf(1.f + 2.f * txy + hx2 * ty2, EPSF);
  float d0 = ((1.f + 2.f * txy + ty2) * nh0 + (1.f - hx2) * ht0) / dden;
  float d1 = ((1.f + 2.f * txy + ty2) * nh1 + (1.f - hx2) * ht1) / dden;

  // pm = mobius_pointwise_mul(z, delta)
  float z0 = zbuf[(long)b * H + i0], z1 = zbuf[(long)b * H + i1];
  float dn2 = block_sum(d0 * d0 + d1 * d1);
  float dn  = fmaxf(sqrtf(dn2 + 1e-30f), EPSF);
  float ld  = atanhf(clipf(dn, -MAXT, MAXT)) / dn;
  float v0 = z0 * ld * d0, v1 = z1 * ld * d1;
  float nv2 = block_sum(v0 * v0 + v1 * v1);
  float nv  = fmaxf(sqrtf(nv2 + 1e-30f), EPSF);
  float se  = tanhf(nv) / nv;
  float p0 = se * v0, p1 = se * v1;

  // h_new = mobius_add(h, pm)
  float hh2 = block_sum(h0 * h0 + h1 * h1);
  float pp2 = block_sum(p0 * p0 + p1 * p1);
  float hp  = block_sum(h0 * p0 + h1 * p1);
  float fden = fmaxf(1.f + 2.f * hp + hh2 * pp2, EPSF);
  float hn0 = ((1.f + 2.f * hp + pp2) * h0 + (1.f - hh2) * p0) / fden;
  float hn1 = ((1.f + 2.f * hp + pp2) * h1 + (1.f - hh2) * p1) / fden;

  hbuf[(long)b * H + i0] = hn0; hbuf[(long)b * H + i1] = hn1;
  h16[(long)b * H + i0] = (_Float16)hn0;
  h16[(long)b * H + i1] = (_Float16)hn1;
  out[((long)b * S + s) * H + i0] = hn0;
  out[((long)b * S + s) * H + i1] = hn1;
  float n2 = block_sum(hn0 * hn0 + hn1 * hn1);
  if (t == 0) hnorm[b] = sqrtf(n2 + 1e-30f);
}

// ---------------------------------------------------------------- launcher
extern "C" void kernel_launch(void* const* d_in, const int* in_sizes, int n_in,
                              void* d_out, int out_size, void* d_ws, size_t ws_size,
                              hipStream_t stream) {
  const int*   ctx  = (const int*)d_in[0];
  const float* lut  = (const float*)d_in[1];
  const float* wihF = (const float*)d_in[2];
  const float* whhF = (const float*)d_in[3];
  const float* bF   = (const float*)d_in[4];
  const float* wihB = (const float*)d_in[5];
  const float* whhB = (const float*)d_in[6];
  const float* bB   = (const float*)d_in[7];
  float* out_f = (float*)d_out;
  float* out_b = out_f + (long)B * S * H;

  char* ws = (char*)d_ws;
  auto alloc = [&](size_t bytes) -> char* {
    char* p = ws; ws += (bytes + 255) & ~(size_t)255; return p;
  };
  _Float16* xh16  = (_Float16*)alloc((size_t)B * S * D * 2);
  float*    xnorm = (float*)   alloc((size_t)B * S * 4);
  _Float16* wih16[2] = {(_Float16*)alloc((size_t)3 * H * D * 2),
                        (_Float16*)alloc((size_t)3 * H * D * 2)};
  _Float16* whh16[2] = {(_Float16*)alloc((size_t)3 * H * H * 2),
                        (_Float16*)alloc((size_t)3 * H * H * 2)};
  float* UX[2] = {(float*)alloc((size_t)B * S * 3 * H * 4),
                  (float*)alloc((size_t)B * S * 3 * H * 4)};
  float* hbuf[2]; _Float16* h16[2]; float* hn[2];
  float* zb[2]; _Float16* rh16[2]; float* rhn[2]; float* whraw[2];
  for (int d = 0; d < 2; ++d) {
    hbuf[d]  = (float*)   alloc((size_t)B * H * 4);
    h16[d]   = (_Float16*)alloc((size_t)B * H * 2);
    hn[d]    = (float*)   alloc((size_t)B * 4);
    zb[d]    = (float*)   alloc((size_t)B * H * 4);
    rh16[d]  = (_Float16*)alloc((size_t)B * H * 2);
    rhn[d]   = (float*)   alloc((size_t)B * 4);
    whraw[d] = (float*)   alloc((size_t)B * 3 * H * 4);
  }
  const float* wih[2] = {wihF, wihB};
  const float* whh[2] = {whhF, whhB};
  const float* bias[2] = {bF, bB};
  float* outp[2] = {out_f, out_b};

  // ---- setup (parallel, off critical path)
  gather_embed<<<B * S, 256, 0, stream>>>(ctx, lut, xh16, xnorm);
  const int NW = 3 * H * D;
  for (int d = 0; d < 2; ++d) {
    f32_to_f16<<<(NW + 255) / 256, 256, 0, stream>>>(wih[d], wih16[d], NW);
    f32_to_f16<<<(NW + 255) / 256, 256, 0, stream>>>(whh[d], whh16[d], NW);
    init_state<<<(B * H + 255) / 256, 256, 0, stream>>>(hbuf[d], h16[d], hn[d]);
  }
  // ---- big input-side GEMMs: UX[d] = xh16 @ w_ih[d]^T  (M=32768,N=1536,K=512)
  {
    dim3 g(3 * H / 16, (B * S) / 64);
    wmma_gemm<<<g, 128, 0, stream>>>(xh16, D, wih16[0], D, UX[0], 3 * H, D);
    wmma_gemm<<<g, 128, 0, stream>>>(xh16, D, wih16[1], D, UX[1], 3 * H, D);
  }
  // ---- sequential scans (two independent directions)
  dim3 gs(H / 16, B / 64);  // per-step GEMM: M=256, N=512
  for (int d = 0; d < 2; ++d) {
    for (int t = 0; t < S; ++t) {
      const int s = (d == 0) ? t : (S - 1 - t);  // input & output position
      // r gate raw: h @ W_hr^T   (rows [0,H) of w_hh)
      wmma_gemm<<<gs, 128, 0, stream>>>(h16[d], H, whh16[d], H,
                                        whraw[d], 3 * H, H);
      // z gate raw: h @ W_hz^T   (rows [2H,3H))
      wmma_gemm<<<gs, 128, 0, stream>>>(h16[d], H, whh16[d] + (size_t)2 * H * H, H,
                                        whraw[d] + 2 * H, 3 * H, H);
      cell_gates<<<B, 256, 0, stream>>>(hbuf[d], hn[d], whraw[d], UX[d], xnorm,
                                        bias[d], s, zb[d], rh16[d], rhn[d]);
      // h_tilde raw: rh @ W_hhh^T (rows [H,2H))
      wmma_gemm<<<gs, 128, 0, stream>>>(rh16[d], H, whh16[d] + (size_t)H * H, H,
                                        whraw[d] + H, 3 * H, H);
      cell_update<<<B, 256, 0, stream>>>(hbuf[d], hn[d], whraw[d], UX[d], xnorm,
                                         bias[d], zb[d], rhn[d], s, h16[d], outp[d]);
    }
  }
}
